// FPModule_38336878084339
// MI455X (gfx1250) — matrix operand order repaired and checked by reading, not compiled, not run on hardware
//
#include <hip/hip_runtime.h>
#include <hip/hip_bf16.h>

#ifndef __has_builtin
#define __has_builtin(x) 0
#endif

typedef __attribute__((ext_vector_type(16))) _Float16 v16h;
typedef __attribute__((ext_vector_type(8)))  _Float16 v8h;
typedef __attribute__((ext_vector_type(4)))  _Float16 v4h;
typedef __attribute__((ext_vector_type(8)))  float    v8f;
typedef __attribute__((ext_vector_type(4)))  unsigned int u32x4;
typedef __attribute__((ext_vector_type(8)))  int      i32x8;
typedef __attribute__((ext_vector_type(4)))  int      i32x4;

// Address-space-qualified int4 pointers for the async global->LDS builtin:
// prototype is (V4i AS1* src, V4i AS3* dst, imm offset, imm cpol).
typedef __attribute__((address_space(1))) i32x4 as1_i32x4;
typedef __attribute__((address_space(3))) i32x4 as3_i32x4;

#define B_      16
#define NX_     1024
#define NY_     4096
#define C_      256
#define CS_     128
#define H_      256
#define KIN_    384            // C_ + CS_
#define M_TOT   (B_ * NY_)     // 65536
#define EPS_    1e-8f

#if defined(__AMDGCN__) && __has_builtin(__builtin_amdgcn_tensor_load_to_lds)
#define HAS_TDM 1
#else
#define HAS_TDM 0
#endif

#if defined(__AMDGCN__) && __has_builtin(__builtin_amdgcn_global_load_async_to_lds_b128)
#define HAS_ASYNC_LDS 1
#else
#define HAS_ASYNC_LDS 0
#endif

// ---------------------------------------------------------------------------
// Kernel 0: transpose + convert weights to f16, [K][N] f32 -> [N][K] f16
// so the WMMA B-operand per lane is a contiguous K-run.
// ---------------------------------------------------------------------------
__global__ void __launch_bounds__(256) prep_weights_kernel(
    const float* __restrict__ W1, const float* __restrict__ W2,
    _Float16* __restrict__ Wt1, _Float16* __restrict__ Wt2) {
  int t = blockIdx.x * 256 + threadIdx.x;
  if (t < KIN_ * H_) {
    const int k = t / H_, n = t % H_;
    Wt1[(size_t)n * KIN_ + k] = (_Float16)W1[t];
  }
  t -= KIN_ * H_;
  if (t >= 0 && t < H_ * H_) {
    const int k = t / H_, n = t % H_;
    Wt2[(size_t)n * H_ + k] = (_Float16)W2[t];
  }
}

// ---------------------------------------------------------------------------
// Kernel 1: k=3 nearest neighbors per fine point.
// One block = (batch, 256-point chunk). Coarse positions (12KB) staged into
// LDS with GLOBAL_LOAD_ASYNC_TO_LDS_B128 (ASYNCcnt path) when available.
// ---------------------------------------------------------------------------
__global__ void __launch_bounds__(256) knn_kernel(
    const float* __restrict__ pos, const float* __restrict__ pos_skip,
    float* __restrict__ wn, int* __restrict__ idx) {
  __shared__ alignas(16) float sp[NX_ * 3];
  const int b     = blockIdx.x >> 4;   // 16 chunks per batch
  const int chunk = blockIdx.x & 15;
  const float* px = pos + (size_t)b * NX_ * 3;

#if HAS_ASYNC_LDS
  for (int i = threadIdx.x; i < (NX_ * 3) / 4; i += 256) {   // 768 x 16B chunks
    __builtin_amdgcn_global_load_async_to_lds_b128(
        (as1_i32x4*)(px + i * 4),      // global source (AS1)
        (as3_i32x4*)(sp + i * 4),      // LDS destination (AS3)
        0, 0);
  }
#if __has_builtin(__builtin_amdgcn_s_wait_asynccnt)
  __builtin_amdgcn_s_wait_asynccnt(0);
#else
  asm volatile("s_wait_asynccnt 0" ::: "memory");
#endif
#else
  for (int i = threadIdx.x; i < NX_ * 3; i += 256) sp[i] = px[i];
#endif
  __syncthreads();

  const int r  = b * NY_ + chunk * 256 + threadIdx.x;
  const float qx = pos_skip[3 * r + 0];
  const float qy = pos_skip[3 * r + 1];
  const float qz = pos_skip[3 * r + 2];

  float d0 = 3.4e38f, d1 = 3.4e38f, d2 = 3.4e38f;
  int   i0 = 0, i1 = 0, i2 = 0;
#pragma unroll 4
  for (int j = 0; j < NX_; ++j) {
    const float dx = sp[3 * j + 0] - qx;
    const float dy = sp[3 * j + 1] - qy;
    const float dz = sp[3 * j + 2] - qz;
    const float d  = dx * dx + dy * dy + dz * dz;
    if (d < d2) {
      if (d < d1) {
        d2 = d1; i2 = i1;
        if (d < d0) { d1 = d0; i1 = i0; d0 = d; i0 = j; }
        else        { d1 = d;  i1 = j; }
      } else { d2 = d; i2 = j; }
    }
  }
  const float w0 = 1.0f / (sqrtf(d0) + EPS_);
  const float w1 = 1.0f / (sqrtf(d1) + EPS_);
  const float w2 = 1.0f / (sqrtf(d2) + EPS_);
  const float s  = 1.0f / (w0 + w1 + w2 + EPS_);
  wn[3 * r + 0] = w0 * s;  wn[3 * r + 1] = w1 * s;  wn[3 * r + 2] = w2 * s;
  idx[3 * r + 0] = b * NX_ + i0;
  idx[3 * r + 1] = b * NX_ + i1;
  idx[3 * r + 2] = b * NX_ + i2;
}

// ---------------------------------------------------------------------------
// Kernel 2: weighted 3-row gather + skip concat -> A1 [M_TOT][384] f16.
// One wave per fine row; float4 loads, v4h (8B) stores.
// ---------------------------------------------------------------------------
__global__ void __launch_bounds__(256) gather_kernel(
    const float* __restrict__ x, const float* __restrict__ x_skip,
    const float* __restrict__ wn, const int* __restrict__ idx,
    _Float16* __restrict__ A1) {
  const int r    = blockIdx.x * 8 + (threadIdx.x >> 5);
  const int lane = threadIdx.x & 31;
  const float w0 = wn[3 * r + 0], w1 = wn[3 * r + 1], w2 = wn[3 * r + 2];
  const float4* f0 = (const float4*)(x + (size_t)idx[3 * r + 0] * C_);
  const float4* f1 = (const float4*)(x + (size_t)idx[3 * r + 1] * C_);
  const float4* f2 = (const float4*)(x + (size_t)idx[3 * r + 2] * C_);
  _Float16* ar = A1 + (size_t)r * KIN_;

#pragma unroll
  for (int c4 = lane; c4 < C_ / 4; c4 += 32) {
    const float4 a = f0[c4], b = f1[c4], c = f2[c4];
    v4h o;
    o[0] = (_Float16)(w0 * a.x + w1 * b.x + w2 * c.x);
    o[1] = (_Float16)(w0 * a.y + w1 * b.y + w2 * c.y);
    o[2] = (_Float16)(w0 * a.z + w1 * b.z + w2 * c.z);
    o[3] = (_Float16)(w0 * a.w + w1 * b.w + w2 * c.w);
    *(v4h*)(ar + c4 * 4) = o;
  }
  {
    const float4 a = ((const float4*)(x_skip + (size_t)r * CS_))[lane];
    v4h o;
    o[0] = (_Float16)a.x; o[1] = (_Float16)a.y;
    o[2] = (_Float16)a.z; o[3] = (_Float16)a.w;
    *(v4h*)(ar + C_ + lane * 4) = o;
  }
}

// ---------------------------------------------------------------------------
// WMMA operand loaders (row-major [row][K] f16 source, global or LDS).
// A 16x32 f16 k order used by this code: lane<16 holds K{0..15}, lane>=16 K{16..31}.
// B 32x16 f16 k order used by this code: lane<16 holds K{0..15}, lane>=16 K{16..31}.
// ---------------------------------------------------------------------------
__device__ __forceinline__ v16h load_opA(const _Float16* p) {
  const v8h lo = *(const v8h*)(p);
  const v8h hi = *(const v8h*)(p + 8);
  return __builtin_shufflevector(lo, hi, 0,1,2,3,4,5,6,7,8,9,10,11,12,13,14,15);
}
__device__ __forceinline__ v16h load_opB(const _Float16* p) {
  const v8h lo = *(const v8h*)(p);
  const v8h hi = *(const v8h*)(p + 8);
  return __builtin_shufflevector(lo, hi, 0,1,2,3,4,5,6,7,8,9,10,11,12,13,14,15);
}

// ---------------------------------------------------------------------------
// Kernels 3/4: GEMM + bias + ReLU via v_wmma_f32_16x16x32_f16.
// Block = 8 waves, 128x128 tile; wave = 32x64 subtile (2x4 WMMA tiles).
// B sub-panel (128 rows x 32 K) staged per k-step into LDS by the Tensor
// Data Mover, double buffered, TDM pad feature gives 80B row pitch.
// ---------------------------------------------------------------------------
#define BPITCH 40   // halves per LDS tile row (32 data + 8 pad)

template<int K, bool OUT_F16>
__global__ void __launch_bounds__(256) gemm_bias_relu_kernel(
    const _Float16* __restrict__ A,    // [M][K] row-major f16
    const _Float16* __restrict__ Bt,   // [256][K] row-major f16 (W^T)
    const float* __restrict__ bias,    // [256]
    _Float16* __restrict__ outH,       // used if OUT_F16
    float* __restrict__ outF) {        // used if !OUT_F16
  __shared__ alignas(16) _Float16 sB[2][128 * BPITCH];   // 2 x 10KB

  const int wave  = threadIdx.x >> 5;
  const int lane  = threadIdx.x & 31;
  const int wm    = wave & 3;                       // 4 M-groups
  const int wnid  = wave >> 2;                      // 2 N-groups
  const int mBase = blockIdx.x * 128 + wm * 32;     // two 16-row tiles
  const int nPanel = blockIdx.y * 128;              // global N base of panel
  const int nTileBase = wnid * 64;                  // local N row in panel
  const int lrow  = lane & 15;
  const int koffA = (lane >> 4) * 16;               // A operand K offset
  const int koffB = (lane >> 4) * 16;               // B operand K offset

  const uint64_t gPanel = (uint64_t)(uintptr_t)(Bt + (size_t)nPanel * K);
  const uint32_t ldsBase = (uint32_t)(uintptr_t)&sB[0][0];
  constexpr uint32_t bufBytes = 128u * BPITCH * 2u;
  constexpr int NK = K / 32;

#if HAS_TDM
  // Build a D# for one 128x32-element f16 tile (row stride K elements) and
  // issue TENSOR_LOAD_TO_LDS. pad: 4 DWORDs after every 16 DWORDs -> 80B pitch.
  auto tdm_issue = [&](int buf, int ks) {
    const uint64_t ga = gPanel + (uint64_t)ks * 64;   // 32 halves = 64B
    u32x4 g0;
    g0[0] = 1u;                                        // count=1, user mode
    g0[1] = ldsBase + (uint32_t)buf * bufBytes;        // lds_addr
    g0[2] = (uint32_t)ga;                              // global_addr[31:0]
    g0[3] = ((uint32_t)(ga >> 32) & 0x01FFFFFFu)       // global_addr[56:32]
            | 0x80000000u;                             // type=2 ("image")
    i32x8 g1;
    g1[0] = (int)((1u << 16)        // data_size = 2 bytes
                | (1u << 20)        // pad_enable
                | (3u << 22)        // pad_interval: 16 DWORDs
                | (3u << 25));      // pad_amount: 4 DWORDs
    g1[1] = (int)((unsigned)K << 16);          // tensor_dim0 = K elems
    g1[2] = (int)(256u << 16);                 // tensor_dim1 = 256 rows
    g1[3] = (int)(32u << 16);                  // tile_dim0 = 32
    g1[4] = 128;                               // tile_dim1 = 128
    g1[5] = K;                                 // tensor_dim0_stride = K
    g1[6] = 0;
    g1[7] = 0;
    const i32x4 z4 = {0, 0, 0, 0};
#if __clang_major__ >= 23
    const i32x8 z8 = {0, 0, 0, 0, 0, 0, 0, 0};
    __builtin_amdgcn_tensor_load_to_lds(g0, g1, z4, z4, z8, 0);
#else
    __builtin_amdgcn_tensor_load_to_lds(g0, g1, z4, z4, 0);
#endif
  };
  auto stage = [&](int buf, int ks) { if (wave == 0) tdm_issue(buf, ks); };
  auto stage_wait = [&]() {
    if (wave == 0) __builtin_amdgcn_s_wait_tensorcnt(0);
  };
#else
  // Fallback: cooperative copy with identical padded LDS layout.
  auto stage = [&](int buf, int ks) {
    const _Float16* src = Bt + (size_t)nPanel * K + ks * 32;
    for (int t = threadIdx.x; t < 128 * 4; t += 256) {
      const int rw = t >> 2, c = (t & 3) * 8;
      *(v8h*)&sB[buf][rw * BPITCH + c] = *(const v8h*)(src + (size_t)rw * K + c);
    }
  };
  auto stage_wait = [&]() {};
#endif

  v8f acc[2][4];
#pragma unroll
  for (int i = 0; i < 2; ++i)
#pragma unroll
    for (int j = 0; j < 4; ++j) acc[i][j] = {};

  const _Float16* a0p = A + (size_t)(mBase + lrow) * K + koffA;
  const _Float16* a1p = a0p + (size_t)16 * K;

  stage(0, 0);
  for (int ks = 0; ks < NK; ++ks) {
    const int buf = ks & 1;
    stage_wait();            // TDM for sB[buf] landed (max 1 outstanding)
    __syncthreads();         // visible to all; readers of sB[buf^1] done
    if (ks + 1 < NK) stage(buf ^ 1, ks + 1);   // overlap DMA with WMMAs

    const int kb = ks * 32;
    const v16h a0 = load_opA(a0p + kb);
    const v16h a1 = load_opA(a1p + kb);
    v16h bt[4];
#pragma unroll
    for (int j = 0; j < 4; ++j)
      bt[j] = load_opB(&sB[buf][(nTileBase + j * 16 + lrow) * BPITCH + koffB]);
#pragma unroll
    for (int j = 0; j < 4; ++j) {
      acc[0][j] = __builtin_amdgcn_wmma_f32_16x16x32_f16(
          false, a0, false, bt[j], (short)0, acc[0][j], false, false);
      acc[1][j] = __builtin_amdgcn_wmma_f32_16x16x32_f16(
          false, a1, false, bt[j], (short)0, acc[1][j], false, false);
    }
  }

  // C/D layout: lanes 0-15 -> rows M=q, lanes 16-31 -> rows M=8+q; N = lane%16
  const int rowHalf = (lane >> 4) * 8;
#pragma unroll
  for (int j = 0; j < 4; ++j) {
    const int col = nPanel + nTileBase + j * 16 + lrow;
    const float bb = bias[col];
#pragma unroll
    for (int i = 0; i < 2; ++i) {
      const int row0 = mBase + i * 16 + rowHalf;
#pragma unroll
      for (int q = 0; q < 8; ++q) {
        float v = acc[i][j][q] + bb;
        v = v > 0.0f ? v : 0.0f;
        if (OUT_F16) outH[(size_t)(row0 + q) * H_ + col] = (_Float16)v;
        else         outF[(size_t)(row0 + q) * H_ + col] = v;
      }
    }
  }
}

// ---------------------------------------------------------------------------
// Host launcher.
// Inputs: 0:x 1:pos 2:batch 3:x_skip 4:pos_skip 5:batch_skip
//         6:W1 7:b1 8:W2 9:b2 10:k   (batch vectors & k unused: fixed layout)
// ---------------------------------------------------------------------------
extern "C" void kernel_launch(void* const* d_in, const int* in_sizes, int n_in,
                              void* d_out, int out_size, void* d_ws, size_t ws_size,
                              hipStream_t stream) {
  const float* x        = (const float*)d_in[0];
  const float* pos      = (const float*)d_in[1];
  const float* x_skip   = (const float*)d_in[3];
  const float* pos_skip = (const float*)d_in[4];
  const float* W1       = (const float*)d_in[6];
  const float* b1       = (const float*)d_in[7];
  const float* W2       = (const float*)d_in[8];
  const float* b2       = (const float*)d_in[9];
  float* out            = (float*)d_out;

  char* ws = (char*)d_ws;
  _Float16* Wt1 = (_Float16*)ws;  ws += (size_t)H_ * KIN_ * sizeof(_Float16);
  _Float16* Wt2 = (_Float16*)ws;  ws += (size_t)H_ * H_   * sizeof(_Float16);
  float*    wn  = (float*)ws;     ws += (size_t)M_TOT * 3 * sizeof(float);
  int*      idx = (int*)ws;       ws += (size_t)M_TOT * 3 * sizeof(int);
  _Float16* A1  = (_Float16*)ws;  ws += (size_t)M_TOT * KIN_ * sizeof(_Float16);
  _Float16* Hb  = (_Float16*)ws;  ws += (size_t)M_TOT * H_   * sizeof(_Float16);

  // 0) weights -> f16 transposed
  prep_weights_kernel<<<(KIN_ * H_ + H_ * H_) / 256, 256, 0, stream>>>(
      W1, W2, Wt1, Wt2);

  // 1) kNN: 16 batches x 16 chunks of 256 fine points
  knn_kernel<<<B_ * (NY_ / 256), 256, 0, stream>>>(pos, pos_skip, wn, idx);

  // 2) interpolate + concat -> A1 f16 (8 rows per block)
  gather_kernel<<<M_TOT / 8, 256, 0, stream>>>(x, x_skip, wn, idx, A1);

  // 3) H = relu(A1 @ W1 + b1), f16 out
  gemm_bias_relu_kernel<KIN_, true><<<dim3(M_TOT / 128, H_ / 128), 256, 0, stream>>>(
      A1, Wt1, b1, Hb, nullptr);

  // 4) out = relu(H @ W2 + b2), f32 out
  gemm_bias_relu_kernel<H_, false><<<dim3(M_TOT / 128, H_ / 128), 256, 0, stream>>>(
      Hb, Wt2, b2, nullptr, out);
}
